// CrossIntrinsicAttnProcessor2_0_61125974556748
// MI455X (gfx1250) — matrix operand order repaired and checked by reading, not compiled, not run on hardware
//
#include <hip/hip_runtime.h>

// ---------------------------------------------------------------------------
// Types
// ---------------------------------------------------------------------------
typedef __bf16 bf16_t;
typedef bf16_t bf16x4 __attribute__((ext_vector_type(4)));
typedef bf16_t v8bf   __attribute__((ext_vector_type(8)));
typedef bf16_t v16bf  __attribute__((ext_vector_type(16)));
typedef float  v8f    __attribute__((ext_vector_type(8)));

#define HD   128
#define NH   24
#define NB   2
#define DD   3072
#define SQ   1536
#define SKL  2560
#define ENCN 512
#define FF   1024
#define MH   (NB * FF)    // 2048 image rows
#define ME   (NB * ENCN)  // 1024 encoder rows

// ---------------------------------------------------------------------------
// WMMA helpers (CDNA5 bf16 16x16x32, f32 accumulate)
// ---------------------------------------------------------------------------
static __device__ __forceinline__ v8f wmma_bf16(v16bf a, v16bf b, v8f c) {
  return __builtin_amdgcn_wmma_f32_16x16x32_bf16(false, a, false, b, (short)0, c,
                                                 false, false);
}

// A-fragment: 16x32 (MxK) bf16, row-major source [*, ld], K-contiguous.
// lane<16 : row = row0+lane,    elems 0..7 -> K k0..k0+7,  8..15 -> k0+16..23
// lane>=16: row = row0+lane-16, elems 0..7 -> K k0+8..15,  8..15 -> k0+24..31
static __device__ __forceinline__ v16bf frag_a(const bf16_t* __restrict__ base,
                                               int ld, int row0, int k0, int lane) {
  int half = lane >> 4, r = lane & 15;
  const bf16_t* p = base + (size_t)(row0 + r) * ld + k0 + half * 8;
  v8bf lo = *(const v8bf*)p;
  v8bf hi = *(const v8bf*)(p + 16);
  v16bf o;
#pragma unroll
  for (int i = 0; i < 8; ++i) { o[i] = lo[i]; o[i + 8] = hi[i]; }
  return o;
}

// B-fragment: 32x16 (KxN) bf16 where source is stored [N, ld] row-major with
// K contiguous (i.e. we multiply by source^T).
// lane<16 : col = col0+lane,    elems 0..7 -> K k0..k0+7,  8..15 -> k0+16..23
// lane>=16: col = col0+lane-16, elems 0..7 -> K k0+8..15,  8..15 -> k0+24..31
static __device__ __forceinline__ v16bf frag_b(const bf16_t* __restrict__ base,
                                               int ld, int col0, int k0, int lane) {
  int half = lane >> 4, r = lane & 15;
  const bf16_t* p = base + (size_t)(col0 + r) * ld + k0 + half * 8;
  v8bf lo = *(const v8bf*)p;
  v8bf hi = *(const v8bf*)(p + 16);
  v16bf o;
#pragma unroll
  for (int i = 0; i < 8; ++i) { o[i] = lo[i]; o[i + 8] = hi[i]; }
  return o;
}

// ---------------------------------------------------------------------------
// fp32 -> bf16 conversion (4-wide)
// ---------------------------------------------------------------------------
__global__ void cvt_bf16_kernel(const float* __restrict__ src,
                                bf16_t* __restrict__ dst, int n4) {
  int i = blockIdx.x * blockDim.x + threadIdx.x;
  if (i < n4) {
    float4 v = ((const float4*)src)[i];
    bf16x4 o = { (bf16_t)v.x, (bf16_t)v.y, (bf16_t)v.z, (bf16_t)v.w };
    ((bf16x4*)dst)[i] = o;
  }
}

// ---------------------------------------------------------------------------
// GEMM: C[M,N] (f32) = A[M,K](bf16) @ W[N,K]^T (bf16) + bias[N]
// block = 256 threads = 8 waves arranged 4(M) x 2(N); block tile 256x128;
// each wave computes a 64x64 tile = 4x4 WMMA subtiles.
// Per 32-wide K-step: 4 A-frags + 4 B-frags (16 b128 loads) feed 16 WMMAs
// -> 1.0 b128 load per WMMA (vs 1.5 with a 64x32 wave tile).
// ---------------------------------------------------------------------------
__global__ __launch_bounds__(256) void gemm_bias_kernel(
    const bf16_t* __restrict__ A, const bf16_t* __restrict__ W,
    const float* __restrict__ bias, float* __restrict__ C, int M, int N, int K) {
  int lane = threadIdx.x & 31;
  int wave = threadIdx.x >> 5;
  int wr = wave >> 1;  // 0..3 (M)
  int wc = wave & 1;   // 0..1 (N)
  int m0 = blockIdx.y * 256 + wr * 64;
  int n0 = blockIdx.x * 128 + wc * 64;

  v8f acc[4][4] = {};
  for (int k0 = 0; k0 < K; k0 += 32) {
    v16bf bfr[4];
#pragma unroll
    for (int j = 0; j < 4; ++j) bfr[j] = frag_b(W, K, n0 + j * 16, k0, lane);
#pragma unroll
    for (int i = 0; i < 4; ++i) {
      v16bf a = frag_a(A, K, m0 + i * 16, k0, lane);
#pragma unroll
      for (int j = 0; j < 4; ++j) acc[i][j] = wmma_bf16(a, bfr[j], acc[i][j]);
    }
  }

  int half = lane >> 4, r = lane & 15;
#pragma unroll
  for (int i = 0; i < 4; ++i)
#pragma unroll
    for (int j = 0; j < 4; ++j) {
      int col = n0 + j * 16 + r;
      float bv = bias ? bias[col] : 0.0f;
#pragma unroll
      for (int p = 0; p < 8; ++p) {
        int row = m0 + i * 16 + p + 8 * half;
        C[(size_t)row * N + col] = acc[i][j][p] + bv;
      }
    }
}

// ---------------------------------------------------------------------------
// RMSNorm + RoPE helper: one wave handles one 128-element head row
// (4 fp32 elements per lane; rope pairs are lane-local).
// ---------------------------------------------------------------------------
static __device__ __forceinline__ void rms_rope4(
    const float* __restrict__ src, const float* __restrict__ w,
    const float* __restrict__ cosb, const float* __restrict__ sinb,
    bool do_norm, bool do_rope, int lane, float out[4]) {
  int o = lane * 4;
  float x0 = src[o + 0], x1 = src[o + 1], x2 = src[o + 2], x3 = src[o + 3];
  if (do_norm) {
    float ss = x0 * x0 + x1 * x1 + x2 * x2 + x3 * x3;
#pragma unroll
    for (int off = 16; off >= 1; off >>= 1) ss += __shfl_xor(ss, off, 32);
    float rr = rsqrtf(ss * (1.0f / 128.0f) + 1e-6f);
    x0 = x0 * rr * w[o + 0];
    x1 = x1 * rr * w[o + 1];
    x2 = x2 * rr * w[o + 2];
    x3 = x3 * rr * w[o + 3];
  }
  if (do_rope) {
    float c0 = cosb[o + 0], c1 = cosb[o + 1], c2 = cosb[o + 2], c3 = cosb[o + 3];
    float s0 = sinb[o + 0], s1 = sinb[o + 1], s2 = sinb[o + 2], s3 = sinb[o + 3];
    out[0] = x0 * c0 - x1 * s0;
    out[1] = x1 * c1 + x0 * s1;
    out[2] = x2 * c2 - x3 * s2;
    out[3] = x3 * c3 + x2 * s3;
  } else {
    out[0] = x0; out[1] = x1; out[2] = x2; out[3] = x3;
  }
}

// Build Q [B,NH,SQ,HD] bf16 (enc part rows 0..511, image part 512..1535)
__global__ __launch_bounds__(128) void build_q_kernel(
    const float* __restrict__ qe, const float* __restrict__ qh,
    const float* __restrict__ naq, const float* __restrict__ nq,
    const float* __restrict__ cosb, const float* __restrict__ sinb,
    bf16_t* __restrict__ Qa) {
  int lane = threadIdx.x & 31;
  int t = blockIdx.x * 4 + (threadIdx.x >> 5);  // row id over B*NH*SQ
  int b = t / (NH * SQ);
  int rem = t % (NH * SQ);
  int h = rem / SQ;
  int s = rem % SQ;
  const float* src;
  const float* w;
  if (s < ENCN) { src = qe + ((size_t)(b * ENCN + s)) * DD + h * HD; w = naq; }
  else          { src = qh + ((size_t)(b * FF + (s - ENCN))) * DD + h * HD; w = nq; }
  float y[4];
  rms_rope4(src, w, cosb + (size_t)s * HD, sinb + (size_t)s * HD, true, true, lane, y);
  bf16_t* dst = Qa + (((size_t)(b * NH + h)) * SQ + s) * HD + lane * 4;
  bf16x4 o = { (bf16_t)y[0], (bf16_t)y[1], (bf16_t)y[2], (bf16_t)y[3] };
  *(bf16x4*)dst = o;
}

// Build K [B,NH,SKL,HD] bf16; image part (rows 512..2559) is the cross-batch
// concatenation j = b'*FF + f, rope position 512 + (j mod FF).
__global__ __launch_bounds__(128) void build_k_kernel(
    const float* __restrict__ ke, const float* __restrict__ kh,
    const float* __restrict__ nak, const float* __restrict__ nk,
    const float* __restrict__ cosb, const float* __restrict__ sinb,
    bf16_t* __restrict__ Ka) {
  int lane = threadIdx.x & 31;
  int t = blockIdx.x * 4 + (threadIdx.x >> 5);  // row id over B*NH*SKL
  int b = t / (NH * SKL);
  int rem = t % (NH * SKL);
  int h = rem / SKL;
  int s = rem % SKL;
  const float* src;
  const float* w;
  int pos;
  if (s < ENCN) { src = ke + ((size_t)(b * ENCN + s)) * DD + h * HD; w = nak; pos = s; }
  else {
    int j = s - ENCN;  // 0..2047 over both batches
    src = kh + ((size_t)j) * DD + h * HD;
    w = nk;
    pos = ENCN + (j & (FF - 1));
  }
  float y[4];
  rms_rope4(src, w, cosb + (size_t)pos * HD, sinb + (size_t)pos * HD, true, true, lane, y);
  bf16_t* dst = Ka + (((size_t)(b * NH + h)) * SKL + s) * HD + lane * 4;
  bf16x4 o = { (bf16_t)y[0], (bf16_t)y[1], (bf16_t)y[2], (bf16_t)y[3] };
  *(bf16x4*)dst = o;
}

// Build V transposed: Vt [B,NH,HD,SKL] bf16 so PV B-fragments are contiguous.
__global__ __launch_bounds__(128) void build_v_kernel(
    const float* __restrict__ ve, const float* __restrict__ vh,
    bf16_t* __restrict__ Vt) {
  int lane = threadIdx.x & 31;
  int t = blockIdx.x * 4 + (threadIdx.x >> 5);
  int b = t / (NH * SKL);
  int rem = t % (NH * SKL);
  int h = rem / SKL;
  int s = rem % SKL;
  const float* src;
  if (s < ENCN) src = ve + ((size_t)(b * ENCN + s)) * DD + h * HD;
  else          src = vh + ((size_t)(s - ENCN)) * DD + h * HD;
  bf16_t* base = Vt + ((size_t)(b * NH + h)) * HD * SKL;
#pragma unroll
  for (int j = 0; j < 4; ++j) {
    int d = lane * 4 + j;
    base[(size_t)d * SKL + s] = (bf16_t)src[d];
  }
}

// ---------------------------------------------------------------------------
// Flash attention: block = 128 threads (4 waves); each wave owns a 16-row Q
// tile; streams Sk in 32-key tiles. QK^T and PV use bf16 WMMA; softmax fp32.
// Output written bf16 into enc / img activation buffers for the final GEMMs.
// ---------------------------------------------------------------------------
__global__ __launch_bounds__(128) void attn_kernel(
    const bf16_t* __restrict__ Qa, const bf16_t* __restrict__ Ka,
    const bf16_t* __restrict__ Vt, bf16_t* __restrict__ out_enc,
    bf16_t* __restrict__ out_img) {
  __shared__ bf16_t lds_p[4][16 * 32];  // per-wave 16x32 probability slab
  int lane = threadIdx.x & 31;
  int wave = threadIdx.x >> 5;
  int half = lane >> 4, r = lane & 15;
  int b = blockIdx.z;
  int h = blockIdx.y;
  int q0 = blockIdx.x * 64 + wave * 16;

  const bf16_t* Q  = Qa + ((size_t)(b * NH + h)) * SQ * HD;
  const bf16_t* Kb = Ka + ((size_t)(b * NH + h)) * SKL * HD;
  const bf16_t* Vb = Vt + ((size_t)(b * NH + h)) * HD * SKL;
  bf16_t* lp = lds_p[wave];

  // Q fragments for this 16-row tile: 4 chunks of K=32 along hd.
  v16bf qf[4];
#pragma unroll
  for (int c = 0; c < 4; ++c) qf[c] = frag_a(Q, HD, q0, c * 32, lane);

  v8f o[8] = {};  // 8 hd-chunks of the 16x128 output tile
  float mrow[8], lrow[8];
#pragma unroll
  for (int p = 0; p < 8; ++p) { mrow[p] = -1e30f; lrow[p] = 0.0f; }
  const float scale = 0.08838834764831845f;  // 128^-0.5

  for (int kt = 0; kt < SKL; kt += 32) {
    if (kt + 32 < SKL)
      __builtin_prefetch(Kb + (size_t)(kt + 32) * HD, 0, 1);

    // S tile 16x32 = two 16x16 WMMA accumulations over hd (K=4x32).
    v8f s0 = {}, s1 = {};
#pragma unroll
    for (int c = 0; c < 4; ++c) {
      v16bf kb0 = frag_b(Kb, HD, kt, c * 32, lane);
      v16bf kb1 = frag_b(Kb, HD, kt + 16, c * 32, lane);
      s0 = wmma_bf16(qf[c], kb0, s0);
      s1 = wmma_bf16(qf[c], kb1, s1);
    }

    // Online softmax over the 32 new columns. Row q = q0 + p + 8*half lives
    // entirely within this half's 16 lanes (one column per lane).
    float e0[8], e1[8];
#pragma unroll
    for (int p = 0; p < 8; ++p) {
      float v0 = s0[p] * scale, v1 = s1[p] * scale;
      float mx = fmaxf(v0, v1);
#pragma unroll
      for (int off = 8; off >= 1; off >>= 1) mx = fmaxf(mx, __shfl_xor(mx, off, 32));
      float mnew = fmaxf(mrow[p], mx);
      float corr = __expf(mrow[p] - mnew);
      e0[p] = __expf(v0 - mnew);
      e1[p] = __expf(v1 - mnew);
      float rs = e0[p] + e1[p];
#pragma unroll
      for (int off = 8; off >= 1; off >>= 1) rs += __shfl_xor(rs, off, 32);
      lrow[p] = lrow[p] * corr + rs;
      mrow[p] = mnew;
#pragma unroll
      for (int t = 0; t < 8; ++t) o[t][p] *= corr;
    }

    // Reshape P (16x32) to an A-fragment via per-wave LDS bounce.
#pragma unroll
    for (int p = 0; p < 8; ++p) {
      int row = p + 8 * half;
      lp[row * 32 + r]      = (bf16_t)e0[p];
      lp[row * 32 + 16 + r] = (bf16_t)e1[p];
    }
    asm volatile("s_wait_dscnt 0" ::: "memory");
    v16bf pf;
    {
      const bf16_t* pp = lp + r * 32 + half * 8;
      v8bf lo = *(const v8bf*)pp;
      v8bf hi = *(const v8bf*)(pp + 16);
#pragma unroll
      for (int i = 0; i < 8; ++i) { pf[i] = lo[i]; pf[i + 8] = hi[i]; }
    }
    asm volatile("s_wait_dscnt 0" ::: "memory");

    // O += P(16x32) x V(32x128): 8 WMMAs (one per 16-wide hd chunk).
#pragma unroll
    for (int t = 0; t < 8; ++t) {
      v16bf vf = frag_b(Vb, SKL, t * 16, kt, lane);
      o[t] = wmma_bf16(pf, vf, o[t]);
    }
  }

  // Normalize and write into the projection-input activation buffers.
#pragma unroll
  for (int p = 0; p < 8; ++p) {
    int s = q0 + p + 8 * half;
    float inv = 1.0f / lrow[p];
    bf16_t* dst;
    if (s < ENCN) dst = out_enc + ((size_t)(b * ENCN + s)) * DD + h * HD;
    else          dst = out_img + ((size_t)(b * FF + (s - ENCN))) * DD + h * HD;
#pragma unroll
    for (int t = 0; t < 8; ++t) dst[t * 16 + r] = (bf16_t)(o[t][p] * inv);
  }
}

// ---------------------------------------------------------------------------
// Host launcher
// ---------------------------------------------------------------------------
extern "C" void kernel_launch(void* const* d_in, const int* in_sizes, int n_in,
                              void* d_out, int out_size, void* d_ws, size_t ws_size,
                              hipStream_t stream) {
  (void)in_sizes; (void)n_in; (void)out_size; (void)ws_size;

  const float* hs   = (const float*)d_in[0];
  const float* enc  = (const float*)d_in[1];
  const float* rcos = (const float*)d_in[2];
  const float* rsin = (const float*)d_in[3];
  const float* wq  = (const float*)d_in[4];  const float* bq  = (const float*)d_in[5];
  const float* wk  = (const float*)d_in[6];  const float* bk  = (const float*)d_in[7];
  const float* wv  = (const float*)d_in[8];  const float* bv  = (const float*)d_in[9];
  const float* waq = (const float*)d_in[10]; const float* baq = (const float*)d_in[11];
  const float* wak = (const float*)d_in[12]; const float* bak = (const float*)d_in[13];
  const float* wav = (const float*)d_in[14]; const float* bav = (const float*)d_in[15];
  const float* wo  = (const float*)d_in[16]; const float* bo  = (const float*)d_in[17];
  const float* wao = (const float*)d_in[18]; const float* bao = (const float*)d_in[19];
  const float* nq  = (const float*)d_in[20]; const float* nk  = (const float*)d_in[21];
  const float* naq = (const float*)d_in[22]; const float* nak = (const float*)d_in[23];

  // ---- workspace carve-up -------------------------------------------------
  char* ws = (char*)d_ws;
  size_t off = 0;
  auto alloc = [&](size_t bytes) {
    void* p = ws + off;
    off += (bytes + 255) & ~(size_t)255;
    return p;
  };
  const size_t WSZ = (size_t)DD * DD;  // 3072*3072
  bf16_t* hsb  = (bf16_t*)alloc((size_t)MH * DD * 2);
  bf16_t* encb = (bf16_t*)alloc((size_t)ME * DD * 2);
  bf16_t* wb[8];
  for (int i = 0; i < 8; ++i) wb[i] = (bf16_t*)alloc(WSZ * 2);
  float* qh = (float*)alloc((size_t)MH * DD * 4);
  float* kh = (float*)alloc((size_t)MH * DD * 4);
  float* vh = (float*)alloc((size_t)MH * DD * 4);
  float* qe = (float*)alloc((size_t)ME * DD * 4);
  float* ke = (float*)alloc((size_t)ME * DD * 4);
  float* ve = (float*)alloc((size_t)ME * DD * 4);
  bf16_t* Qa = (bf16_t*)alloc((size_t)NB * NH * SQ * HD * 2);
  bf16_t* Ka = (bf16_t*)alloc((size_t)NB * NH * SKL * HD * 2);
  bf16_t* Vt = (bf16_t*)alloc((size_t)NB * NH * HD * SKL * 2);
  bf16_t* oenc = (bf16_t*)alloc((size_t)ME * DD * 2);
  bf16_t* oimg = (bf16_t*)alloc((size_t)MH * DD * 2);

  // ---- 1) fp32 -> bf16 conversions ---------------------------------------
  auto cvt = [&](const float* s, bf16_t* d, size_t n) {
    int n4 = (int)(n / 4);
    cvt_bf16_kernel<<<(n4 + 255) / 256, 256, 0, stream>>>(s, d, n4);
  };
  cvt(hs, hsb, (size_t)MH * DD);
  cvt(enc, encb, (size_t)ME * DD);
  const float* wsrc[8] = { wq, wk, wv, waq, wak, wav, wo, wao };
  for (int i = 0; i < 8; ++i) cvt(wsrc[i], wb[i], WSZ);

  // ---- 2) QKV projections (bf16 WMMA, f32 out) ----------------------------
  dim3 gH(DD / 128, MH / 256);  // (24,8)
  dim3 gE(DD / 128, ME / 256);  // (24,4)
  gemm_bias_kernel<<<gH, 256, 0, stream>>>(hsb,  wb[0], bq,  qh, MH, DD, DD);
  gemm_bias_kernel<<<gH, 256, 0, stream>>>(hsb,  wb[1], bk,  kh, MH, DD, DD);
  gemm_bias_kernel<<<gH, 256, 0, stream>>>(hsb,  wb[2], bv,  vh, MH, DD, DD);
  gemm_bias_kernel<<<gE, 256, 0, stream>>>(encb, wb[3], baq, qe, ME, DD, DD);
  gemm_bias_kernel<<<gE, 256, 0, stream>>>(encb, wb[4], bak, ke, ME, DD, DD);
  gemm_bias_kernel<<<gE, 256, 0, stream>>>(encb, wb[5], bav, ve, ME, DD, DD);

  // ---- 3) RMSNorm + RoPE + attention layout -------------------------------
  build_q_kernel<<<(NB * NH * SQ) / 4, 128, 0, stream>>>(qe, qh, naq, nq, rcos, rsin, Qa);
  build_k_kernel<<<(NB * NH * SKL) / 4, 128, 0, stream>>>(ke, kh, nak, nk, rcos, rsin, Ka);
  build_v_kernel<<<(NB * NH * SKL) / 4, 128, 0, stream>>>(ve, vh, Vt);

  // ---- 4) attention -------------------------------------------------------
  dim3 gA(SQ / 64, NH, NB);  // (24,24,2)
  attn_kernel<<<gA, 128, 0, stream>>>(Qa, Ka, Vt, oenc, oimg);

  // ---- 5) output projections straight into d_out --------------------------
  float* hs_out  = (float*)d_out;                       // [2,1024,3072]
  float* enc_out = hs_out + (size_t)MH * DD;            // [2,512,3072]
  gemm_bias_kernel<<<gH, 256, 0, stream>>>(oimg, wb[6], bo,  hs_out,  MH, DD, DD);
  gemm_bias_kernel<<<gE, 256, 0, stream>>>(oenc, wb[7], bao, enc_out, ME, DD, DD);
}
